// AMXQuaternionicLinear_2980707303755
// MI455X (gfx1250) — matrix operand order repaired and checked by reading, not compiled, not run on hardware
//
#include <hip/hip_runtime.h>

typedef __attribute__((ext_vector_type(16))) __bf16        v16bf;
typedef __attribute__((ext_vector_type(8)))  float         v8f;
typedef __attribute__((ext_vector_type(4)))  float         v4f;
typedef __attribute__((ext_vector_type(8)))  unsigned int  v8u;

#define IN_F  128
#define OUT_F 128

// Hamilton-product expansion tables: out component c <- input component d uses
// weight component c_comp[c*4+d] with sign c_sign[c*4+d].
__constant__ int   c_comp[16] = {0,1,2,3,  1,0,3,2,  2,3,0,1,  3,2,1,0};
__constant__ float c_sign[16] = {1,-1,-1,-1,  1,1,1,-1,  1,-1,1,1,  1,1,-1,1};

// Pack two floats' top-16 bits (bf16 truncation) into one dword: a -> [15:0], b -> [31:16]
__device__ __forceinline__ unsigned pack_bf(float a, float b) {
    return (__float_as_uint(b) & 0xFFFF0000u) | (__float_as_uint(a) >> 16);
}
__device__ __forceinline__ float trunc_bf(float a) {
    return __uint_as_float(__float_as_uint(a) & 0xFFFF0000u);
}

__global__ __launch_bounds__(256)
void quat_linear_wmma_bf16x3(const float* __restrict__ x,
                             const float* __restrict__ w,     // [32][32][4]
                             const float* __restrict__ bias,  // [128]
                             float* __restrict__ out,         // [rows][128]
                             int rows)
{
    // Expanded real weight, hi/lo bf16 split, stored transposed [n_out][k_in]
    // so each B-fragment lane reads 8 consecutive packed dwords (32B).
    __shared__ __align__(32) unsigned short sWh[OUT_F * IN_F];  // 32 KB
    __shared__ __align__(32) unsigned short sWl[OUT_F * IN_F];  // 32 KB

    for (int e = threadIdx.x; e < OUT_F * IN_F; e += blockDim.x) {
        int nf = e >> 7;          // output feature 0..127
        int kf = e & 127;         // input feature  0..127
        int o  = nf >> 2, c = nf & 3;
        int n  = kf >> 2, d = kf & 3;
        int t  = c * 4 + d;
        float wr = c_sign[t] * w[(o * 32 + n) * 4 + c_comp[t]];
        unsigned b = __float_as_uint(wr);
        sWh[e] = (unsigned short)(b >> 16);
        float lof = wr - __uint_as_float(b & 0xFFFF0000u);
        sWl[e] = (unsigned short)(__float_as_uint(lof) >> 16);
    }
    __syncthreads();

    const int lane = threadIdx.x & 31;
    const int wave = threadIdx.x >> 5;
    const int m    = lane & 15;
    const int half = lane >> 4;
    const long row0 = ((long)blockIdx.x * 8 + wave) * 16;
    if (row0 >= rows) return;

    v8f acc[8];
    #pragma unroll
    for (int t = 0; t < 8; ++t) acc[t] = (v8f){};

    const float* xrow = x + (row0 + m) * (long)IN_F;

    // Rolled K loop: keeps live set < 256 VGPRs (no vgpr-msb switching),
    // one iteration = 4 global b128 + ~60 conversion VALU (co-executes with
    // XDL) + 32 ds b128 + 24 bf16 WMMAs.
    #pragma unroll 1
    for (int kk = 0; kk < IN_F; kk += 32) {
        // ---- A fragments (16-bit A 16x32 layout) ----
        // lane (m, half): VGPR j -> K = kk + 16*half + 2j, +1
        const float* p0 = xrow + kk + 16 * half;
        v4f f0 = *(const v4f*)(p0);        // K offsets 16h+0..3
        v4f f1 = *(const v4f*)(p0 + 4);    // 16h+4..7
        v4f f2 = *(const v4f*)(p0 + 8);    // 16h+8..11
        v4f f3 = *(const v4f*)(p0 + 12);   // 16h+12..15
        float c16[16];
        #pragma unroll
        for (int i = 0; i < 4; ++i) {
            c16[i] = f0[i]; c16[4 + i] = f1[i];
            c16[8 + i] = f2[i]; c16[12 + i] = f3[i];
        }
        v8u ah, al;
        #pragma unroll
        for (int j = 0; j < 8; ++j) {
            float a0 = c16[2 * j], a1 = c16[2 * j + 1];
            ah[j] = pack_bf(a0, a1);
            al[j] = pack_bf(a0 - trunc_bf(a0), a1 - trunc_bf(a1));
        }
        v16bf Ah = __builtin_bit_cast(v16bf, ah);
        v16bf Al = __builtin_bit_cast(v16bf, al);

        // ---- per N-tile: B fragments (dense 32x16 bf16 layout) + 3 WMMAs ----
        // Base address shared by all tiles; tile stride 16*128*2 = 4096 B fits
        // the 16-bit DS immediate offset.
        const unsigned short* wrow = &sWh[m * IN_F + kk + 16 * half];
        #pragma unroll
        for (int t = 0; t < 8; ++t) {
            v16bf Bh = __builtin_bit_cast(v16bf,
                *(const v8u*)(wrow + t * 16 * IN_F));
            v16bf Bl = __builtin_bit_cast(v16bf,
                *(const v8u*)(wrow + t * 16 * IN_F + (OUT_F * IN_F)));  // sWl
            acc[t] = __builtin_amdgcn_wmma_f32_16x16x32_bf16(
                false, Ah, false, Bh, (short)0, acc[t], false, false);
            acc[t] = __builtin_amdgcn_wmma_f32_16x16x32_bf16(
                false, Ah, false, Bl, (short)0, acc[t], false, false);
            acc[t] = __builtin_amdgcn_wmma_f32_16x16x32_bf16(
                false, Al, false, Bh, (short)0, acc[t], false, false);
        }
    }

    // Epilogue: C/D layout — VGPR r = row (row0 + r + 8*half), col = t*16 + m
    #pragma unroll
    for (int t = 0; t < 8; ++t) {
        float bv = bias[t * 16 + m];
        #pragma unroll
        for (int r = 0; r < 8; ++r) {
            long rr = row0 + r + 8 * half;
            out[rr * (long)OUT_F + t * 16 + m] = acc[t][r] + bv;
        }
    }
}

extern "C" void kernel_launch(void* const* d_in, const int* in_sizes, int n_in,
                              void* d_out, int out_size, void* d_ws, size_t ws_size,
                              hipStream_t stream) {
    const float* x    = (const float*)d_in[0];
    const float* w    = (const float*)d_in[1];
    const float* bias = (const float*)d_in[2];
    float* out        = (float*)d_out;

    int rows = in_sizes[0] / IN_F;           // 262144
    int blocks = (rows + 127) / 128;         // 8 waves x 16 rows per block
    quat_linear_wmma_bf16x3<<<blocks, 256, 0, stream>>>(x, w, bias, out, rows);
}
